// HyperNetSIR_41566693490892
// MI455X (gfx1250) — matrix operand order repaired and checked by reading, not compiled, not run on hardware
//
#include <hip/hip_runtime.h>

#define N_NODES 20000
#define N_EDGES 5000
#define STEPS   50

// Padded dims and K-splits
#define EPAD    5008   // edge count padded to multiple of 16 (M tiles for e = H@I)
#define KEPAD   5120   // edge dim padded to 32*CHK2*NSEG2 (K for y = H^T e)
#define NSEG1   25     // K-split segments for e = H@I (625 chunks of 32 / 25)
#define CHK1    25     // k-chunks (32 wide) per segment, first matvec
#define NSEG2   5      // K-split segments for y = H^T e (160 chunks / 5)
#define CHK2    32     // k-chunks per segment, second matvec

typedef __bf16 bf16;
typedef bf16  v8bf  __attribute__((ext_vector_type(8)));
typedef bf16  v16bf __attribute__((ext_vector_type(16)));
typedef float v8f   __attribute__((ext_vector_type(8)));

// ---------------------------------------------------------------------------
// WMMA matvec: column of A(bf16, lda) @ vec(bf16).
// One wave per (M-tile, K-segment); deterministic partials to part[].
//
// A-layout (ISA 7.12.2, 16-bit A 16x32): lane m (hi=0) holds K {0..7,16..23},
// lane m+16 (hi=1) holds K {8..15,24..31} of row m -> two b128 loads/chunk.
// B trick: EVERY lane loads the vector chunk for its half (lane<16 -> K{0..7,16..23},
// lane>=16 -> K{8..15,24..31}), i.e. the vector is replicated into all 16 columns of
// B. Every column of D then equals A@v; we store column 0. This keeps the
// inner loop fully convergent: no EXEC masking, no per-iteration zeroing.
// C/D column 0: lane 0 -> M=0..7 in c[0..7]; lane 16 -> M=8..15.
// ---------------------------------------------------------------------------
__global__ __launch_bounds__(32) void sir_wmma_matvec(
    const bf16* __restrict__ A, int lda,
    const bf16* __restrict__ vec,
    float* __restrict__ part, int part_ld,
    int rows_valid, int kchunks)
{
  const int lane = threadIdx.x;
  const int hi   = lane >> 4;
  const int mt   = blockIdx.x;
  const int seg  = blockIdx.y;
  const long k0  = (long)seg * kchunks * 32;
  const int  row = mt * 16 + (lane & 15);
  const bf16* __restrict__ arow  = A + (long)row * lda + k0 + 8 * hi;
  const bf16* __restrict__ vhalf = vec + k0 + 8 * hi;

  v8f c = {0.f, 0.f, 0.f, 0.f, 0.f, 0.f, 0.f, 0.f};

  #pragma unroll 4
  for (int t = 0; t < kchunks; ++t) {
    const int k = t * 32;
    // A tile: two 16B chunks per lane (K {0..7,16..23} or {8..15,24..31})
    v8bf a0 = *(const v8bf*)(arow + k);
    v8bf a1 = *(const v8bf*)(arow + k + 16);
    v16bf a = __builtin_shufflevector(a0, a1,
        0, 1, 2, 3, 4, 5, 6, 7, 8, 9, 10, 11, 12, 13, 14, 15);

    // B tile: vector chunk broadcast into all 16 columns (uniform per half)
    v8bf b0 = *(const v8bf*)(vhalf + k);
    v8bf b1 = *(const v8bf*)(vhalf + k + 16);
    v16bf b = __builtin_shufflevector(b0, b1,
        0, 1, 2, 3, 4, 5, 6, 7, 8, 9, 10, 11, 12, 13, 14, 15);

    // Prefetch the A stream ~8 chunks (512B/lane) ahead (speculative; OOB ok)
    __builtin_prefetch(arow + k + 256, 0, 0);

    c = __builtin_amdgcn_wmma_f32_16x16x32_bf16(
        /*neg_a=*/false, a, /*neg_b=*/false, b,
        /*c_mod=*/(short)0, c, /*reuse_a=*/false, /*reuse_b=*/false);
  }

  if ((lane & 15) == 0) {
    const int r0 = mt * 16 + 8 * hi;
    float* __restrict__ op = part + (long)seg * part_ld + r0;
    #pragma unroll
    for (int r = 0; r < 8; ++r)
      if (r0 + r < rows_valid) op[r] = c[r];
  }
}

// Deterministic fixed-order reduction of e-partials -> bf16 e (zero padded).
__global__ __launch_bounds__(256) void sir_reduce_e(
    const float* __restrict__ epart, bf16* __restrict__ ebf)
{
  const int i = blockIdx.x * 256 + threadIdx.x;
  if (i >= KEPAD) return;
  float s = 0.f;
  if (i < N_EDGES) {
    #pragma unroll
    for (int g = 0; g < NSEG1; ++g) s += epart[g * EPAD + i];
  }
  ebf[i] = (bf16)s;
}

// Reduce y-partials + SIR state update + write trajectory frame + refresh Ibf.
__global__ __launch_bounds__(256) void sir_update(
    const float* __restrict__ ypart,
    const float* __restrict__ beta, const float* __restrict__ gamma,
    float* __restrict__ state, bf16* __restrict__ ibf,
    float* __restrict__ traj_t)
{
  const int i = blockIdx.x * 256 + threadIdx.x;
  if (i >= N_NODES) return;
  float y = 0.f;
  #pragma unroll
  for (int g = 0; g < NSEG2; ++g) y += ypart[g * N_NODES + i];
  const float S = state[3 * i + 0];
  const float I = state[3 * i + 1];
  const float R = state[3 * i + 2];
  const float nc = beta[i] * S * y;
  const float nr = gamma[i] * I;
  float s0 = fmaxf(S - nc, 0.f);
  float s1 = fmaxf(I + nc - nr, 0.f);
  float s2 = fmaxf(R + nr, 0.f);
  const float inv = 1.f / (s0 + s1 + s2);
  s0 *= inv; s1 *= inv; s2 *= inv;
  state[3 * i + 0] = s0; state[3 * i + 1] = s1; state[3 * i + 2] = s2;
  traj_t[3 * i + 0] = s0; traj_t[3 * i + 1] = s1; traj_t[3 * i + 2] = s2;
  ibf[i] = (bf16)s1;
}

// Initial state: frame 0 = x, state = x, Ibf = bf16(x[:,1]).
__global__ __launch_bounds__(256) void sir_init(
    const float* __restrict__ x, float* __restrict__ state,
    bf16* __restrict__ ibf, float* __restrict__ traj0)
{
  const int i = blockIdx.x * 256 + threadIdx.x;
  if (i >= N_NODES) return;
  const float s0 = x[3 * i + 0], s1 = x[3 * i + 1], s2 = x[3 * i + 2];
  state[3 * i + 0] = s0; state[3 * i + 1] = s1; state[3 * i + 2] = s2;
  traj0[3 * i + 0] = s0; traj0[3 * i + 1] = s1; traj0[3 * i + 2] = s2;
  ibf[i] = (bf16)s1;
}

// Once per launch: H (f32, row-major) -> bf16 row-major copy, padded rows zero.
__global__ __launch_bounds__(256) void sir_conv_hrow(
    const float* __restrict__ H, bf16* __restrict__ Hrow)
{
  const long idx = (long)blockIdx.x * 256 + threadIdx.x;
  if (idx >= (long)EPAD * N_NODES) return;
  const long e = idx / N_NODES;
  Hrow[idx] = (e < N_EDGES) ? (bf16)H[idx] : (bf16)0.0f;
}

// Once per launch: H (f32) -> bf16 transposed (Hcol[i][j] = H[j][i]),
// LDS-tiled so both global read and global write stay coalesced.
__global__ __launch_bounds__(256) void sir_conv_hcol(
    const float* __restrict__ H, bf16* __restrict__ Hcol)
{
  __shared__ float tile[32][33];
  const int tx  = threadIdx.x;        // 0..31
  const int ty0 = threadIdx.y;        // 0..7
  const int i0 = blockIdx.x * 32;     // node dim (exactly 20000/32 blocks)
  const int j0 = blockIdx.y * 32;     // edge dim (padded to 5120)
  #pragma unroll
  for (int r = 0; r < 4; ++r) {
    const int ty = ty0 + r * 8;
    const int j = j0 + ty;
    tile[ty][tx] = (j < N_EDGES) ? H[(long)j * N_NODES + i0 + tx] : 0.f;
  }
  __syncthreads();
  #pragma unroll
  for (int r = 0; r < 4; ++r) {
    const int ty = ty0 + r * 8;
    Hcol[(long)(i0 + ty) * KEPAD + j0 + tx] = (bf16)tile[tx][ty];
  }
}

extern "C" void kernel_launch(void* const* d_in, const int* in_sizes, int n_in,
                              void* d_out, int out_size, void* d_ws, size_t ws_size,
                              hipStream_t stream)
{
  (void)in_sizes; (void)n_in; (void)out_size; (void)ws_size;
  const float* x     = (const float*)d_in[0];
  const float* H     = (const float*)d_in[1];
  const float* beta  = (const float*)d_in[2];
  const float* gamma = (const float*)d_in[3];
  float* traj = (float*)d_out;

  // Workspace carve-out (~406 MB)
  char* w = (char*)d_ws;
  auto carve = [&w](size_t bytes) {
    char* p = w;
    w += (bytes + 255) & ~(size_t)255;
    return p;
  };
  bf16*  Hrow  = (bf16*) carve((size_t)EPAD * N_NODES * sizeof(bf16));
  bf16*  Hcol  = (bf16*) carve((size_t)N_NODES * KEPAD * sizeof(bf16));
  float* state = (float*)carve((size_t)N_NODES * 3 * sizeof(float));
  bf16*  ibf   = (bf16*) carve((size_t)N_NODES * sizeof(bf16));
  bf16*  ebf   = (bf16*) carve((size_t)KEPAD * sizeof(bf16));
  float* epart = (float*)carve((size_t)NSEG1 * EPAD * sizeof(float));
  float* ypart = (float*)carve((size_t)NSEG2 * N_NODES * sizeof(float));

  // One-time conversions + init
  {
    const long total = (long)EPAD * N_NODES;           // 100,160,000 -> 391,250 blocks
    sir_conv_hrow<<<dim3((unsigned)((total + 255) / 256)), dim3(256), 0, stream>>>(H, Hrow);
  }
  sir_conv_hcol<<<dim3(N_NODES / 32, KEPAD / 32), dim3(32, 8), 0, stream>>>(H, Hcol);
  sir_init<<<dim3((N_NODES + 255) / 256), dim3(256), 0, stream>>>(x, state, ibf, traj);

  // 49 sequential steps
  for (int t = 1; t < STEPS; ++t) {
    // e = H @ I   (M = 5008 padded rows, K = 20000)
    sir_wmma_matvec<<<dim3(EPAD / 16, NSEG1), dim3(32), 0, stream>>>(
        Hrow, N_NODES, ibf, epart, EPAD, N_EDGES, CHK1);
    sir_reduce_e<<<dim3(KEPAD / 256), dim3(256), 0, stream>>>(epart, ebf);
    // y = H^T @ e (M = 20000, K = 5120 padded)
    sir_wmma_matvec<<<dim3(N_NODES / 16, NSEG2), dim3(32), 0, stream>>>(
        Hcol, KEPAD, ebf, ypart, N_NODES, N_NODES, CHK2);
    sir_update<<<dim3((N_NODES + 255) / 256), dim3(256), 0, stream>>>(
        ypart, beta, gamma, state, ibf, traj + (size_t)t * N_NODES * 3);
  }
}
